// VisionTransformerMoE_62148176773290
// MI455X (gfx1250) — matrix operand order repaired and checked by reading, not compiled, not run on hardware
//
#include <hip/hip_runtime.h>

// ---------------------------------------------------------------------------
// Types for WMMA fragments (gfx1250, wave32)
// ---------------------------------------------------------------------------
typedef _Float16 h8_t  __attribute__((ext_vector_type(8)));
typedef _Float16 h16_t __attribute__((ext_vector_type(16)));
typedef float    f8_t  __attribute__((ext_vector_type(8)));

#define T_TOK   3152      // 16 * 197
#define NPATCH  196
#define TPATCH  3136      // 16 * 196
#define DMODEL  384
#define HFFN    1536
#define NEXP    8
#define NSLOT   6304      // 2 * T_TOK

__device__ __forceinline__ float gelu_exact(float x) {
    return 0.5f * x * (1.0f + erff(x * 0.70710678118654752f));
}

// 32-byte global->LDS async copy (2 x b128; INST_OFFSET shifts both addresses)
__device__ __forceinline__ void async_copy32B(void* lds_dst, const void* gsrc) {
    unsigned lds = (unsigned)(unsigned long long)lds_dst;  // low 32b = LDS offset
    asm volatile(
        "global_load_async_to_lds_b128 %0, %1, off\n\t"
        "global_load_async_to_lds_b128 %0, %1, off offset:16"
        :: "v"(lds), "v"(gsrc) : "memory");
}
__device__ __forceinline__ void wait_async_pipe(bool has_next) {
    if (has_next) asm volatile("s_wait_asynccnt 4" ::: "memory");
    else          asm volatile("s_wait_asynccnt 0" ::: "memory");
}

// ---------------------------------------------------------------------------
// Batched transpose + f32->f16 convert:  in[b][K][N] f32  ->  out[b][N][K] f16
// ---------------------------------------------------------------------------
__global__ __launch_bounds__(256)
void transpose_cvt_kernel(const float* __restrict__ in, _Float16* __restrict__ out,
                          int K, int N, long stride_in, long stride_out)
{
    __shared__ float tile[32][33];
    const float* ip = in + (long)blockIdx.z * stride_in;
    _Float16*    op = out + (long)blockIdx.z * stride_out;
    const int n0 = blockIdx.x * 32, k0 = blockIdx.y * 32;
#pragma unroll
    for (int i = 0; i < 4; ++i) {
        int k = k0 + threadIdx.y + i * 8, n = n0 + threadIdx.x;
        tile[threadIdx.y + i * 8][threadIdx.x] = (k < K && n < N) ? ip[(long)k * N + n] : 0.0f;
    }
    __syncthreads();
#pragma unroll
    for (int i = 0; i < 4; ++i) {
        int n = n0 + threadIdx.y + i * 8, k = k0 + threadIdx.x;
        if (n < N && k < K)
            op[(long)n * K + k] = (_Float16)tile[threadIdx.x][threadIdx.y + i * 8];
    }
}

// ---------------------------------------------------------------------------
// GEMM: C[M,N] (op)= A[M,K] @ W[K,N] + bias
//   A: f16 [M][lda], Wt: f16 [N][K].  Both staged to LDS with async b128
//   copies, double-buffered; clamped addressing (no predication: garbage
//   rows/cols only feed unstored C rows/cols).  K % 32 == 0.
//   act: 0 none, 1 GELU.  accum: 0 store f32, 1 add f32.  outf16: store f16.
// ---------------------------------------------------------------------------
__global__ __launch_bounds__(128)
void gemm_kernel(const _Float16* __restrict__ A, int lda,
                 const _Float16* __restrict__ Wt,
                 const float* __restrict__ bias,
                 void* __restrict__ Cv, int ldc,
                 int M, int N, int K, int act, int accum, int outf16)
{
    __shared__ __attribute__((aligned(16))) _Float16 As[2][64][40];
    __shared__ __attribute__((aligned(16))) _Float16 Bs[2][64][40];

    const int tid   = threadIdx.x;
    const int mbase = blockIdx.x * 64;
    const int nbase = blockIdx.y * 64;
    const int wave  = tid >> 5;
    const int lane  = tid & 31;
    const int warpM = (wave >> 1) * 32;
    const int warpN = (wave & 1) * 32;
    const int lrow  = lane & 15;
    const int lhi   = lane >> 4;

    // staging assignment: one thread = 16 contiguous halves of one row
    const int rl  = tid >> 1;
    const int c0  = (tid & 1) * 16;
    const int arow = min(mbase + rl, M - 1);   // clamp: safe, stores guarded
    const int bcol = min(nbase + rl, N - 1);
    const _Float16* aptr = A + (long)arow * lda + c0;
    const _Float16* bptr = Wt + (long)bcol * K + c0;

    f8_t acc[2][2] = {};
    const int nstage = K / 32;

    // prologue: stage 0
    async_copy32B(&As[0][rl][c0], aptr);
    async_copy32B(&Bs[0][rl][c0], bptr);

    for (int it = 0; it < nstage; ++it) {
        const int buf = it & 1;
        const bool has_next = (it + 1 < nstage);
        if (has_next) {
            int k1 = (it + 1) * 32;
            async_copy32B(&As[buf ^ 1][rl][c0], aptr + k1);
            async_copy32B(&Bs[buf ^ 1][rl][c0], bptr + k1);
        }
        wait_async_pipe(has_next);
        __syncthreads();

        h16_t afrag[2], bfrag[2];
#pragma unroll
        for (int mt = 0; mt < 2; ++mt) {
            int am = warpM + mt * 16 + lrow;
            h8_t lo = *(const h8_t*)&As[buf][am][lhi * 8];
            h8_t hi = *(const h8_t*)&As[buf][am][lhi * 8 + 16];
            afrag[mt] = __builtin_shufflevector(lo, hi, 0,1,2,3,4,5,6,7,8,9,10,11,12,13,14,15);
        }
#pragma unroll
        for (int nt = 0; nt < 2; ++nt) {
            int bn = warpN + nt * 16 + lrow;
            h8_t lo = *(const h8_t*)&Bs[buf][bn][lhi * 8];
            h8_t hi = *(const h8_t*)&Bs[buf][bn][lhi * 8 + 16];
            bfrag[nt] = __builtin_shufflevector(lo, hi, 0,1,2,3,4,5,6,7,8,9,10,11,12,13,14,15);
        }
#pragma unroll
        for (int mt = 0; mt < 2; ++mt)
#pragma unroll
            for (int nt = 0; nt < 2; ++nt)
                acc[mt][nt] = __builtin_amdgcn_wmma_f32_16x16x32_f16(
                    false, afrag[mt], false, bfrag[nt],
                    (short)0, acc[mt][nt], false, false);
        __syncthreads();
    }

#pragma unroll
    for (int mt = 0; mt < 2; ++mt)
#pragma unroll
        for (int nt = 0; nt < 2; ++nt) {
            int col = nbase + warpN + nt * 16 + lrow;
            if (col >= N) continue;
            float bv = bias ? bias[col] : 0.0f;
#pragma unroll
            for (int r = 0; r < 8; ++r) {
                int row = mbase + warpM + mt * 16 + r + 8 * lhi;
                if (row >= M) continue;
                float v = acc[mt][nt][r] + bv;
                if (act) v = gelu_exact(v);
                long o = (long)row * ldc + col;
                if (outf16) ((_Float16*)Cv)[o] = (_Float16)v;
                else {
                    float* C = (float*)Cv;
                    if (accum) C[o] += v; else C[o] = v;
                }
            }
        }
}

// ---------------------------------------------------------------------------
// MoE expert GEMM: expert e = blockIdx.z; rows = tokens assigned to e.
//   Y[slot,N] = gelu?(X[row(slot),K] @ W_e + b_e); row = perm[slot] or slot.
// ---------------------------------------------------------------------------
__global__ __launch_bounds__(128)
void moe_gemm_kernel(const _Float16* __restrict__ X, int ldx,
                     const _Float16* __restrict__ Wall,
                     const float* __restrict__ ball,
                     void* __restrict__ Yv, int ldy,
                     const int* __restrict__ offs,
                     const int* __restrict__ perm, // null => identity (slot)
                     int K, int N, int act, int outf16)
{
    const int e   = blockIdx.z;
    const int s0  = offs[e];
    const int cnt = offs[e + 1] - s0;
    const int mbase = blockIdx.x * 64;
    if (mbase >= cnt) return;

    const _Float16* Wt  = Wall + (long)e * K * N;
    const float*   bias = ball + (long)e * N;
    const int nbase = blockIdx.y * 64;

    __shared__ __attribute__((aligned(16))) _Float16 As[2][64][40];
    __shared__ __attribute__((aligned(16))) _Float16 Bs[2][64][40];

    const int tid  = threadIdx.x;
    const int wave = tid >> 5;
    const int lane = tid & 31;
    const int warpM = (wave >> 1) * 32;
    const int warpN = (wave & 1) * 32;
    const int lrow = lane & 15;
    const int lhi  = lane >> 4;

    const int rl  = tid >> 1;
    const int c0  = (tid & 1) * 16;
    const int rloc = min(mbase + rl, cnt - 1);
    const int arow = perm ? perm[s0 + rloc] : (s0 + rloc);
    const int bcol = min(nbase + rl, N - 1);
    const _Float16* aptr = X + (long)arow * ldx + c0;
    const _Float16* bptr = Wt + (long)bcol * K + c0;

    f8_t acc[2][2] = {};
    const int nstage = K / 32;

    async_copy32B(&As[0][rl][c0], aptr);
    async_copy32B(&Bs[0][rl][c0], bptr);

    for (int it = 0; it < nstage; ++it) {
        const int buf = it & 1;
        const bool has_next = (it + 1 < nstage);
        if (has_next) {
            int k1 = (it + 1) * 32;
            async_copy32B(&As[buf ^ 1][rl][c0], aptr + k1);
            async_copy32B(&Bs[buf ^ 1][rl][c0], bptr + k1);
        }
        wait_async_pipe(has_next);
        __syncthreads();

        h16_t afrag[2], bfrag[2];
#pragma unroll
        for (int mt = 0; mt < 2; ++mt) {
            int am = warpM + mt * 16 + lrow;
            h8_t lo = *(const h8_t*)&As[buf][am][lhi * 8];
            h8_t hi = *(const h8_t*)&As[buf][am][lhi * 8 + 16];
            afrag[mt] = __builtin_shufflevector(lo, hi, 0,1,2,3,4,5,6,7,8,9,10,11,12,13,14,15);
        }
#pragma unroll
        for (int nt = 0; nt < 2; ++nt) {
            int bn = warpN + nt * 16 + lrow;
            h8_t lo = *(const h8_t*)&Bs[buf][bn][lhi * 8];
            h8_t hi = *(const h8_t*)&Bs[buf][bn][lhi * 8 + 16];
            bfrag[nt] = __builtin_shufflevector(lo, hi, 0,1,2,3,4,5,6,7,8,9,10,11,12,13,14,15);
        }
#pragma unroll
        for (int mt = 0; mt < 2; ++mt)
#pragma unroll
            for (int nt = 0; nt < 2; ++nt)
                acc[mt][nt] = __builtin_amdgcn_wmma_f32_16x16x32_f16(
                    false, afrag[mt], false, bfrag[nt],
                    (short)0, acc[mt][nt], false, false);
        __syncthreads();
    }

#pragma unroll
    for (int mt = 0; mt < 2; ++mt)
#pragma unroll
        for (int nt = 0; nt < 2; ++nt) {
            int col = nbase + warpN + nt * 16 + lrow;
            if (col >= N) continue;
            float bv = bias[col];
#pragma unroll
            for (int r = 0; r < 8; ++r) {
                int rr = mbase + warpM + mt * 16 + r + 8 * lhi;
                if (rr >= cnt) continue;
                float v = acc[mt][nt][r] + bv;
                if (act) v = gelu_exact(v);
                long o = (long)(s0 + rr) * ldy + col;
                if (outf16) ((_Float16*)Yv)[o] = (_Float16)v;
                else        ((float*)Yv)[o] = v;
            }
        }
}

// ---------------------------------------------------------------------------
// Patchify: [B,3,224,224] f32 -> [3136, 768] f16 (k = c*256 + py*16 + px)
// ---------------------------------------------------------------------------
__global__ void patchify_kernel(const float* __restrict__ x, _Float16* __restrict__ patches)
{
    long idx = (long)blockIdx.x * blockDim.x + threadIdx.x;
    if (idx >= (long)TPATCH * 768) return;
    int tp = (int)(idx / 768), k = (int)(idx % 768);
    int b = tp / NPATCH, p = tp % NPATCH;
    int gy = p / 14, gx = p % 14;
    int c = k / 256, rem = k % 256, py = rem / 16, px = rem % 16;
    patches[idx] = (_Float16)x[(((long)(b * 3 + c) * 224 + gy * 16 + py) * 224) + gx * 16 + px];
}

// t[b, n, d] = (n==0 ? cls[d] : patch_out[b*196 + n-1, d]) + pos[n, d]
__global__ void assemble_kernel(const float* __restrict__ patch_out,
                                const float* __restrict__ cls,
                                const float* __restrict__ pos,
                                float* __restrict__ t)
{
    long idx = (long)blockIdx.x * blockDim.x + threadIdx.x;
    if (idx >= (long)T_TOK * DMODEL) return;
    int d = (int)(idx % DMODEL);
    long rn = idx / DMODEL;
    int n = (int)(rn % 197), b = (int)(rn / 197);
    float v = (n == 0) ? cls[d] : patch_out[((long)b * NPATCH + (n - 1)) * DMODEL + d];
    t[idx] = v + pos[(long)n * DMODEL + d];
}

// ---------------------------------------------------------------------------
// LayerNorm over D=384 -> f16 output (feeds WMMA GEMMs)
// ---------------------------------------------------------------------------
__global__ __launch_bounds__(128)
void ln_kernel(const float* __restrict__ x, const float* __restrict__ w,
               const float* __restrict__ b, _Float16* __restrict__ y)
{
    __shared__ float red[128];
    const int row = blockIdx.x, tid = threadIdx.x;
    const float* xr = x + (long)row * DMODEL;

    float s = 0.f;
    for (int d = tid; d < DMODEL; d += 128) s += xr[d];
    red[tid] = s; __syncthreads();
    for (int st = 64; st > 0; st >>= 1) { if (tid < st) red[tid] += red[tid + st]; __syncthreads(); }
    float mu = red[0] * (1.0f / DMODEL);
    __syncthreads();

    float v = 0.f;
    for (int d = tid; d < DMODEL; d += 128) { float t = xr[d] - mu; v += t * t; }
    red[tid] = v; __syncthreads();
    for (int st = 64; st > 0; st >>= 1) { if (tid < st) red[tid] += red[tid + st]; __syncthreads(); }
    float inv = rsqrtf(red[0] * (1.0f / DMODEL) + 1e-6f);

    for (int d = tid; d < DMODEL; d += 128)
        y[(long)row * DMODEL + d] = (_Float16)((xr[d] - mu) * inv * w[d] + b[d]);
}

// ---------------------------------------------------------------------------
// Fused attention; qkv f32 [T,1152]; output f16 [T,384]
// ---------------------------------------------------------------------------
__global__ __launch_bounds__(128)
void attn_kernel(const float* __restrict__ qkv, _Float16* __restrict__ out)
{
    const int i = blockIdx.x, h = blockIdx.y, b = blockIdx.z;
    const int tid = threadIdx.x;
    __shared__ float qs[64];
    __shared__ float sc[197];
    __shared__ float red[128];

    const long base = (long)b * 197 * 1152;
    const long rowi = base + (long)i * 1152;
    if (tid < 64) qs[tid] = qkv[rowi + h * 64 + tid];
    __syncthreads();

    const float scale = 0.125f; // 64^-0.5
    for (int m = tid; m < 197; m += 128) {
        const float* kp = &qkv[base + (long)m * 1152 + 384 + h * 64];
        float d = 0.f;
#pragma unroll 8
        for (int dd = 0; dd < 64; ++dd) d += qs[dd] * kp[dd];
        sc[m] = d * scale;
    }
    __syncthreads();

    float mx = -1e30f;
    for (int m = tid; m < 197; m += 128) mx = fmaxf(mx, sc[m]);
    red[tid] = mx; __syncthreads();
    for (int st = 64; st > 0; st >>= 1) { if (tid < st) red[tid] = fmaxf(red[tid], red[tid + st]); __syncthreads(); }
    mx = red[0];
    __syncthreads();

    float sm = 0.f;
    for (int m = tid; m < 197; m += 128) { float e = expf(sc[m] - mx); sc[m] = e; sm += e; }
    red[tid] = sm; __syncthreads();
    for (int st = 64; st > 0; st >>= 1) { if (tid < st) red[tid] += red[tid + st]; __syncthreads(); }
    float inv = 1.0f / red[0];

    if (tid < 64) {
        const float* vp = &qkv[base + 768 + h * 64 + tid];
        float acc = 0.f;
        for (int m = 0; m < 197; ++m) acc += sc[m] * vp[(long)m * 1152];
        out[((long)b * 197 + i) * DMODEL + h * 64 + tid] = (_Float16)(acc * inv);
    }
}

// ---------------------------------------------------------------------------
// Gate: logits = y(f16) @ gw [384,8]; top-2 + softmax.
// ---------------------------------------------------------------------------
__global__ void gate_topk_kernel(const _Float16* __restrict__ y, const float* __restrict__ gw,
                                 int* __restrict__ topi, float* __restrict__ topg)
{
    int t = blockIdx.x * blockDim.x + threadIdx.x;
    if (t >= T_TOK) return;
    const _Float16* xr = y + (long)t * DMODEL;
    float lg[NEXP];
#pragma unroll
    for (int e = 0; e < NEXP; ++e) lg[e] = 0.f;
    for (int d = 0; d < DMODEL; ++d) {
        float xv = (float)xr[d];
#pragma unroll
        for (int e = 0; e < NEXP; ++e) lg[e] += xv * gw[d * NEXP + e];
    }
    int e0 = 0; float v0 = lg[0];
#pragma unroll
    for (int e = 1; e < NEXP; ++e) if (lg[e] > v0) { v0 = lg[e]; e0 = e; }
    int e1 = -1; float v1 = -1e30f;
#pragma unroll
    for (int e = 0; e < NEXP; ++e) if (e != e0 && lg[e] > v1) { v1 = lg[e]; e1 = e; }
    float ex = expf(v1 - v0);
    float g0 = 1.0f / (1.0f + ex);
    float g1 = ex * g0;
    topi[t * 2] = e0; topi[t * 2 + 1] = e1;
    topg[t * 2] = g0; topg[t * 2 + 1] = g1;
}

// Single-block routing: counts -> offsets -> perm/slot_of (integer atomics)
__global__ void route_kernel(const int* __restrict__ topi, int* __restrict__ offs,
                             int* __restrict__ perm, int* __restrict__ slot_of)
{
    __shared__ int cnt[NEXP], cur[NEXP], off_s[NEXP + 1];
    const int tid = threadIdx.x;
    if (tid < NEXP) cnt[tid] = 0;
    __syncthreads();
    for (int s = tid; s < NSLOT; s += 256) atomicAdd(&cnt[topi[s]], 1);
    __syncthreads();
    if (tid == 0) {
        int a = 0;
        for (int e = 0; e < NEXP; ++e) { off_s[e] = a; a += cnt[e]; }
        off_s[NEXP] = a;
        for (int e = 0; e < NEXP; ++e) cur[e] = off_s[e];
    }
    __syncthreads();
    if (tid <= NEXP) offs[tid] = off_s[tid];
    for (int s = tid; s < NSLOT; s += 256) {
        int e = topi[s];
        int p = atomicAdd(&cur[e], 1);
        perm[p] = s >> 1;       // token id
        slot_of[s] = p;
    }
}

// t[tok] += g0 * Y2[slot0] + g1 * Y2[slot1]   (no float atomics -> deterministic)
__global__ void moe_combine_kernel(float* __restrict__ t, const float* __restrict__ Y2,
                                   const int* __restrict__ slot_of, const float* __restrict__ topg)
{
    long idx = (long)blockIdx.x * blockDim.x + threadIdx.x;
    if (idx >= (long)T_TOK * DMODEL) return;
    int tok = (int)(idx / DMODEL), d = (int)(idx % DMODEL);
    int s0 = slot_of[tok * 2], s1 = slot_of[tok * 2 + 1];
    float g0 = topg[tok * 2], g1 = topg[tok * 2 + 1];
    t[idx] += g0 * Y2[(long)s0 * DMODEL + d] + g1 * Y2[(long)s1 * DMODEL + d];
}

// ---------------------------------------------------------------------------
// Host orchestration
// ---------------------------------------------------------------------------
static inline int cdiv(int a, int b) { return (a + b - 1) / b; }

extern "C" void kernel_launch(void* const* d_in, const int* in_sizes, int n_in,
                              void* d_out, int out_size, void* d_ws, size_t ws_size,
                              hipStream_t stream)
{
    (void)in_sizes; (void)n_in; (void)out_size; (void)ws_size;
    const float* x        = (const float*)d_in[0];
    const float* patch_w  = (const float*)d_in[1];
    const float* patch_b  = (const float*)d_in[2];
    const float* cls_tok  = (const float*)d_in[3];
    const float* pos_emb  = (const float*)d_in[4];
    const float* ln1_w    = (const float*)d_in[5];
    const float* ln1_b    = (const float*)d_in[6];
    const float* qkv_w    = (const float*)d_in[7];
    const float* qkv_b    = (const float*)d_in[8];
    const float* proj_w   = (const float*)d_in[9];
    const float* proj_b   = (const float*)d_in[10];
    const float* ln2_w    = (const float*)d_in[11];
    const float* ln2_b    = (const float*)d_in[12];
    const float* fc1_w    = (const float*)d_in[13];
    const float* fc1_b    = (const float*)d_in[14];
    const float* fc2_w    = (const float*)d_in[15];
    const float* fc2_b    = (const float*)d_in[16];
    const float* gate_w   = (const float*)d_in[17];
    const float* moe_w1   = (const float*)d_in[18];
    const float* moe_b1   = (const float*)d_in[19];
    const float* moe_w2   = (const float*)d_in[20];
    const float* moe_b2   = (const float*)d_in[21];
    const float* norm_w   = (const float*)d_in[22];
    const float* norm_b   = (const float*)d_in[23];
    const float* head_w   = (const float*)d_in[24];
    const float* head_b   = (const float*)d_in[25];

    // ---- byte-based workspace layout (256B aligned slices) ----
    char* base = (char*)d_ws;
    size_t cur = 0;
    auto alloc = [&](size_t bytes) { char* p = base + cur; cur += (bytes + 255) & ~(size_t)255; return p; };

    float*     t_buf   = (float*)    alloc((size_t)T_TOK * DMODEL * 4);
    _Float16*  y16     = (_Float16*) alloc((size_t)T_TOK * DMODEL * 2);
    float*     qkv_buf = (float*)    alloc((size_t)T_TOK * 1152 * 4);
    _Float16*  at16    = (_Float16*) alloc((size_t)T_TOK * DMODEL * 2);
    _Float16*  h16     = (_Float16*) alloc((size_t)NSLOT * HFFN * 2);
    float*     ym_buf  = (float*)    alloc((size_t)NSLOT * DMODEL * 4);
    _Float16*  patches = (_Float16*) alloc((size_t)TPATCH * 768 * 2);
    float*     patch_o = (float*)    alloc((size_t)TPATCH * DMODEL * 4);
    float*     topg    = (float*)    alloc((size_t)NSLOT * 4);
    int*       topi    = (int*)      alloc((size_t)NSLOT * 4);
    int*       perm    = (int*)      alloc((size_t)NSLOT * 4);
    int*       slot_of = (int*)      alloc((size_t)NSLOT * 4);
    int*       offs    = (int*)      alloc(16 * 4);

    // f16 transposed weights (Wt[N][K])
    _Float16* wt_patch = (_Float16*) alloc((size_t)768 * DMODEL * 2);
    _Float16* wt_qkv   = (_Float16*) alloc((size_t)8 * DMODEL * 1152 * 2);
    _Float16* wt_proj  = (_Float16*) alloc((size_t)8 * DMODEL * DMODEL * 2);
    _Float16* wt_fc1   = (_Float16*) alloc((size_t)4 * DMODEL * HFFN * 2);
    _Float16* wt_fc2   = (_Float16*) alloc((size_t)4 * HFFN * DMODEL * 2);
    _Float16* wt_m1    = (_Float16*) alloc((size_t)32 * DMODEL * HFFN * 2);
    _Float16* wt_m2    = (_Float16*) alloc((size_t)32 * HFFN * DMODEL * 2);
    _Float16* wt_head  = (_Float16*) alloc((size_t)DMODEL * 1000 * 2);

    // ---- weight convert+transpose (once per launch, ~12us of HBM traffic) ----
    {
        dim3 blk(32, 8);
        auto tl = [&](const float* in, _Float16* out, int K, int N, int batch) {
            dim3 g(cdiv(N, 32), cdiv(K, 32), batch);
            transpose_cvt_kernel<<<g, blk, 0, stream>>>(in, out, K, N, (long)K * N, (long)K * N);
        };
        tl(patch_w, wt_patch, 768, DMODEL, 1);
        tl(qkv_w,   wt_qkv,   DMODEL, 1152, 8);
        tl(proj_w,  wt_proj,  DMODEL, DMODEL, 8);
        tl(fc1_w,   wt_fc1,   DMODEL, HFFN, 4);
        tl(fc2_w,   wt_fc2,   HFFN, DMODEL, 4);
        tl(moe_w1,  wt_m1,    DMODEL, HFFN, 32);
        tl(moe_w2,  wt_m2,    HFFN, DMODEL, 32);
        tl(head_w,  wt_head,  DMODEL, 1000, 1);
    }

    // ---- patch embedding ----
    {
        long n = (long)TPATCH * 768;
        patchify_kernel<<<(unsigned)((n + 255) / 256), 256, 0, stream>>>(x, patches);
        dim3 g(cdiv(TPATCH, 64), cdiv(DMODEL, 64));
        gemm_kernel<<<g, 128, 0, stream>>>(patches, 768, wt_patch, patch_b,
                                           patch_o, DMODEL, TPATCH, DMODEL, 768, 0, 0, 0);
        long n2 = (long)T_TOK * DMODEL;
        assemble_kernel<<<(unsigned)((n2 + 255) / 256), 256, 0, stream>>>(patch_o, cls_tok, pos_emb, t_buf);
    }

    for (int i = 0; i < 8; ++i) {
        const int j = i / 2;
        // --- attention block ---
        ln_kernel<<<T_TOK, 128, 0, stream>>>(t_buf, ln1_w + i * DMODEL, ln1_b + i * DMODEL, y16);
        {
            dim3 g(cdiv(T_TOK, 64), cdiv(1152, 64));
            gemm_kernel<<<g, 128, 0, stream>>>(y16, DMODEL, wt_qkv + (size_t)i * DMODEL * 1152,
                                               qkv_b + i * 1152, qkv_buf, 1152,
                                               T_TOK, 1152, DMODEL, 0, 0, 0);
        }
        attn_kernel<<<dim3(197, 6, 16), 128, 0, stream>>>(qkv_buf, at16);
        {
            dim3 g(cdiv(T_TOK, 64), cdiv(DMODEL, 64));
            gemm_kernel<<<g, 128, 0, stream>>>(at16, DMODEL, wt_proj + (size_t)i * DMODEL * DMODEL,
                                               proj_b + i * DMODEL, t_buf, DMODEL,
                                               T_TOK, DMODEL, DMODEL, 0, 1, 0); // += residual
        }
        // --- FFN block ---
        ln_kernel<<<T_TOK, 128, 0, stream>>>(t_buf, ln2_w + i * DMODEL, ln2_b + i * DMODEL, y16);
        if ((i & 1) == 0) {
            dim3 g1(cdiv(T_TOK, 64), cdiv(HFFN, 64));
            gemm_kernel<<<g1, 128, 0, stream>>>(y16, DMODEL, wt_fc1 + (size_t)j * DMODEL * HFFN,
                                                fc1_b + j * HFFN, h16, HFFN,
                                                T_TOK, HFFN, DMODEL, 1, 0, 1); // GELU -> f16
            dim3 g2(cdiv(T_TOK, 64), cdiv(DMODEL, 64));
            gemm_kernel<<<g2, 128, 0, stream>>>(h16, HFFN, wt_fc2 + (size_t)j * HFFN * DMODEL,
                                                fc2_b + j * DMODEL, t_buf, DMODEL,
                                                T_TOK, DMODEL, HFFN, 0, 1, 0); // += residual
        } else {
            gate_topk_kernel<<<cdiv(T_TOK, 256), 256, 0, stream>>>(y16, gate_w + (size_t)j * DMODEL * NEXP,
                                                                   topi, topg);
            route_kernel<<<1, 256, 0, stream>>>(topi, offs, perm, slot_of);
            {
                dim3 g1(cdiv(NSLOT, 64), cdiv(HFFN, 64), NEXP);
                moe_gemm_kernel<<<g1, 128, 0, stream>>>(y16, DMODEL,
                                                        wt_m1 + (size_t)j * NEXP * DMODEL * HFFN,
                                                        moe_b1 + (size_t)j * NEXP * HFFN,
                                                        h16, HFFN, offs, perm, DMODEL, HFFN, 1, 1);
                dim3 g2(cdiv(NSLOT, 64), cdiv(DMODEL, 64), NEXP);
                moe_gemm_kernel<<<g2, 128, 0, stream>>>(h16, HFFN,
                                                        wt_m2 + (size_t)j * NEXP * HFFN * DMODEL,
                                                        moe_b2 + (size_t)j * NEXP * DMODEL,
                                                        ym_buf, DMODEL, offs, nullptr, HFFN, DMODEL, 0, 0);
            }
            long n = (long)T_TOK * DMODEL;
            moe_combine_kernel<<<(unsigned)((n + 255) / 256), 256, 0, stream>>>(t_buf, ym_buf, slot_of, topg);
        }
    }

    // ---- final norm + classifier head (CLS rows strided via lda trick) ----
    ln_kernel<<<T_TOK, 128, 0, stream>>>(t_buf, norm_w, norm_b, y16);
    {
        dim3 g(cdiv(16, 64), cdiv(1000, 64));
        gemm_kernel<<<g, 128, 0, stream>>>(y16, 197 * DMODEL, wt_head, head_b,
                                           (float*)d_out, 1000, 16, 1000, DMODEL, 0, 0, 0);
    }
}
